// MoELayer_31499290149013
// MI455X (gfx1250) — matrix operand order repaired and checked by reading, not compiled, not run on hardware
//
#include <hip/hip_runtime.h>
#include <hip/hip_bf16.h>
#include <math.h>

// ---------------------------------------------------------------------------
// MoE layer, MI455X (gfx1250, wave32, WMMA bf16 16x16x32).
// Sparse top-2 routing (4x FLOP cut vs dense all-expert).
// Weights are converted fp32->bf16 AND transposed once per call so both WMMA
// operands read contraction-contiguous data; hot-loop LDS staging is pure
// b128 copies (async-to-LDS, double buffered).
// ---------------------------------------------------------------------------

typedef __attribute__((ext_vector_type(16))) __bf16 v16bf;
typedef __attribute__((ext_vector_type(8)))  __bf16 v8bf;
typedef __attribute__((ext_vector_type(4)))  __bf16 v4bf;
typedef __attribute__((ext_vector_type(8)))  float  v8f;
typedef __attribute__((ext_vector_type(4)))  int    v4i;

#define B_TOK 8192
#define DDIM  1024
#define HDIM  4096
#define NEXP  8

#define BM 64
#define BN 128
#define KC 64
#define LDSS (KC + 8)   // padded stride (bf16); row stride 144B -> conflict-free b128

#define GLOBAL_AS __attribute__((address_space(1)))
#define LDS_AS    __attribute__((address_space(3)))

#if defined(__has_builtin)
#  if __has_builtin(__builtin_amdgcn_global_load_async_to_lds_b128)
#    define HAVE_ASYNC_LDS 1
#  endif
#endif
#ifndef HAVE_ASYNC_LDS
#  define HAVE_ASYNC_LDS 0
#endif

__device__ __forceinline__ void copy16(const __bf16* g, __bf16* l) {
#if HAVE_ASYNC_LDS
  __builtin_amdgcn_global_load_async_to_lds_b128(
      (GLOBAL_AS v4i*)g, (LDS_AS v4i*)l, 0, 0);
#else
  *(v8bf*)l = *(const v8bf*)g;
#endif
}

__device__ __forceinline__ void wait_async() {
#if HAVE_ASYNC_LDS
#  if __has_builtin(__builtin_amdgcn_s_wait_asynccnt)
  __builtin_amdgcn_s_wait_asynccnt(0);
#  else
  asm volatile("s_wait_asynccnt 0x0" ::: "memory");
#  endif
#endif
}

__device__ __forceinline__ v16bf concat8(v8bf lo, v8bf hi) {
  return __builtin_shufflevector(lo, hi, 0,1,2,3,4,5,6,7,8,9,10,11,12,13,14,15);
}

// ------------------------------ fp32 -> bf16 -------------------------------
__global__ void cvt_bf16_k(const float* __restrict__ src, __bf16* __restrict__ dst) {
  size_t i = ((size_t)blockIdx.x * blockDim.x + threadIdx.x) * 4;
  float4 v = *(const float4*)(src + i);
  v4bf o = { (__bf16)v.x, (__bf16)v.y, (__bf16)v.z, (__bf16)v.w };
  *(v4bf*)(dst + i) = o;
}

// ------------------ fp32 -> bf16 with transpose (per expert) ---------------
// src: [E][R][C] f32 ; dst: [E][C][R] bf16. 64x64 tiles through LDS.
__global__ void cvt_tr_k(const float* __restrict__ src, __bf16* __restrict__ dst,
                         int R, int C) {
  int e = blockIdx.z;
  src += (size_t)e * R * C;
  dst += (size_t)e * R * C;
  int c0 = blockIdx.x * 64, r0 = blockIdx.y * 64;
  __shared__ __align__(16) __bf16 tile[64][72];
  int tid = threadIdx.x;
#pragma unroll
  for (int it = 0; it < 4; ++it) {
    int u = tid + it * 256;
    int rr = u >> 4, c4 = u & 15;
    float4 v = *(const float4*)(src + (size_t)(r0 + rr) * C + c0 + c4 * 4);
    tile[rr][c4 * 4 + 0] = (__bf16)v.x;
    tile[rr][c4 * 4 + 1] = (__bf16)v.y;
    tile[rr][c4 * 4 + 2] = (__bf16)v.z;
    tile[rr][c4 * 4 + 3] = (__bf16)v.w;
  }
  __syncthreads();
#pragma unroll
  for (int it = 0; it < 2; ++it) {
    int u = tid + it * 256;
    int cc = u >> 3, rs = u & 7;
    v8bf o;
#pragma unroll
    for (int j = 0; j < 8; ++j) o[j] = tile[rs * 8 + j][cc];
    *(v8bf*)(dst + (size_t)(c0 + cc) * R + r0 + rs * 8) = o;
  }
}

// ------------------------------- zero row ----------------------------------
__global__ void zero_bf16_k(__bf16* __restrict__ p) {
  v8bf z = {};
  *(v8bf*)(p + (size_t)(blockIdx.x * blockDim.x + threadIdx.x) * 8) = z;
}

// ------------------------------- gating ------------------------------------
__global__ void gating_k(const float* __restrict__ x, const float* __restrict__ Wg,
                         const float* __restrict__ bg,
                         int* __restrict__ topk_idx, float* __restrict__ topk_w,
                         float* __restrict__ partial) {
  int wv   = threadIdx.x >> 5;
  int lane = threadIdx.x & 31;
  int t    = blockIdx.x * 8 + wv;

  float acc[NEXP];
#pragma unroll
  for (int e = 0; e < NEXP; ++e) acc[e] = 0.f;

  const float* xr = x + (size_t)t * DDIM;
  for (int d = lane; d < DDIM; d += 32) {
    float xv = xr[d];
    const float* w = Wg + (size_t)d * NEXP;
#pragma unroll
    for (int e = 0; e < NEXP; ++e) acc[e] += xv * w[e];
  }
#pragma unroll
  for (int off = 16; off > 0; off >>= 1) {
#pragma unroll
    for (int e = 0; e < NEXP; ++e) acc[e] += __shfl_xor(acc[e], off, 32);
  }

  __shared__ float probs_s[8][NEXP];
  if (lane == 0) {
    float p[NEXP];
    float mx = -1e30f;
#pragma unroll
    for (int e = 0; e < NEXP; ++e) { p[e] = acc[e] + bg[e]; mx = fmaxf(mx, p[e]); }
    float s = 0.f;
#pragma unroll
    for (int e = 0; e < NEXP; ++e) { p[e] = expf(p[e] - mx); s += p[e]; }
    float inv = 1.f / s;
#pragma unroll
    for (int e = 0; e < NEXP; ++e) p[e] *= inv;
    // top-2, first-index tie break (matches lax.top_k)
    int i0 = 0;
#pragma unroll
    for (int e = 1; e < NEXP; ++e) if (p[e] > p[i0]) i0 = e;
    int i1 = -1; float bv = -1.f;
#pragma unroll
    for (int e = 0; e < NEXP; ++e) if (e != i0 && p[e] > bv) { bv = p[e]; i1 = e; }
    float w0 = p[i0], w1 = p[i1], sw = w0 + w1;
    topk_idx[t * 2 + 0] = i0;  topk_idx[t * 2 + 1] = i1;
    topk_w  [t * 2 + 0] = w0 / sw;  topk_w[t * 2 + 1] = w1 / sw;
#pragma unroll
    for (int e = 0; e < NEXP; ++e) probs_s[wv][e] = p[e];
  }
  __syncthreads();
  if (threadIdx.x < NEXP) {
    float s = 0.f;
#pragma unroll
    for (int w = 0; w < 8; ++w) s += probs_s[w][threadIdx.x];
    partial[blockIdx.x * NEXP + threadIdx.x] = s;
  }
}

// ----------------------------- aux loss ------------------------------------
__global__ void aux_k(const float* __restrict__ partial, int nblk, float* __restrict__ aux_out) {
  __shared__ float sh[NEXP];
  if (threadIdx.x < NEXP) {
    float s = 0.f;
    for (int b = 0; b < nblk; ++b) s += partial[b * NEXP + threadIdx.x];
    float u = s / (float)B_TOK;
    sh[threadIdx.x] = u * logf(u + 1e-8f);
  }
  __syncthreads();
  if (threadIdx.x == 0) {
    float a = 0.f;
#pragma unroll
    for (int e = 0; e < NEXP; ++e) a += sh[e];
    aux_out[0] = 0.01f * a;
  }
}

// ------------------- deterministic per-expert compaction -------------------
__global__ void compact_k(const int* __restrict__ topk_idx, const float* __restrict__ topk_w,
                          int* __restrict__ lists, float* __restrict__ wlists,
                          int* __restrict__ rowmap, int* __restrict__ counts) {
  int e = blockIdx.x;
  __shared__ int wave_tot[8];
  __shared__ int wave_base[8];
  __shared__ int base_s;
  if (threadIdx.x == 0) base_s = 0;
  __syncthreads();
  int lane = threadIdx.x & 31, wv = threadIdx.x >> 5;
  for (int c0 = 0; c0 < B_TOK; c0 += 256) {
    int t = c0 + threadIdx.x;
    int k = -1;
    if (topk_idx[t * 2 + 0] == e) k = 0;
    else if (topk_idx[t * 2 + 1] == e) k = 1;
    unsigned long long mask = __ballot(k >= 0);
    int prefix = __popcll(mask & ((1ull << lane) - 1ull));
    if (lane == 0) wave_tot[wv] = __popcll(mask);
    __syncthreads();
    if (threadIdx.x == 0) {
      int b = base_s;
#pragma unroll
      for (int w = 0; w < 8; ++w) { wave_base[w] = b; b += wave_tot[w]; }
      base_s = b;
    }
    __syncthreads();
    if (k >= 0) {
      int pos = wave_base[wv] + prefix;
      lists [e * B_TOK + pos] = t;
      wlists[e * B_TOK + pos] = topk_w[t * 2 + k];
      rowmap[t * 2 + k] = pos;
    }
    __syncthreads();
  }
  if (threadIdx.x == 0) counts[e] = base_s;
}

__global__ void offsets_k(const int* __restrict__ counts, int* __restrict__ offs) {
  if (threadIdx.x == 0) {
    int s = 0;
#pragma unroll
    for (int e = 0; e < NEXP; ++e) { offs[e] = s; s += counts[e]; }
  }
}

// ---------------- GEMM1: H = relu(X_gathered @ W1_e + b1) ------------------
// W1T: [E][H][D] (pre-transposed). Tile 64x128, 8 waves (2x4), wave 32x32.
__global__ void __launch_bounds__(256)
ffn1_k(const __bf16* __restrict__ xb, const __bf16* __restrict__ W1T,
       const float* __restrict__ b1, const __bf16* __restrict__ zrow,
       const int* __restrict__ lists, const int* __restrict__ counts,
       const int* __restrict__ offs, __bf16* __restrict__ Hb) {
  int e  = blockIdx.z;
  int n0 = blockIdx.x * BN;
  int m0 = blockIdx.y * BM;
  int cnt = counts[e];
  if (m0 >= cnt) return;
  int gbase = offs[e];
  const __bf16* Wt = W1T + (size_t)e * DDIM * HDIM;

  __shared__ __align__(16) __bf16 As[2][BM * LDSS];
  __shared__ __align__(16) __bf16 Bs[2][BN * LDSS];
  __shared__ const __bf16* rowptr[BM];

  int tid = threadIdx.x;
  if (tid < BM) {
    int lr = m0 + tid;
    rowptr[tid] = (lr < cnt) ? (xb + (size_t)lists[e * B_TOK + lr] * DDIM) : zrow;
  }
  __syncthreads();

  int wave = tid >> 5, lane = tid & 31;
  int wm = wave >> 2, wn = wave & 3;
  int laneLow = lane & 15, hiSel = lane >> 4;

  auto stage = [&](int k0, int buf) {
#pragma unroll
    for (int it = 0; it < 2; ++it) {
      int u = tid + it * 256;
      int r = u >> 3, s = u & 7;
      copy16(rowptr[r] + k0 + s * 8, &As[buf][r * LDSS + s * 8]);
    }
#pragma unroll
    for (int it = 0; it < 4; ++it) {
      int u = tid + it * 256;
      int nn = u >> 3, ss = u & 7;
      copy16(Wt + (size_t)(n0 + nn) * DDIM + k0 + ss * 8, &Bs[buf][nn * LDSS + ss * 8]);
    }
  };

  v8f acc[2][2] = {};
  stage(0, 0);
  const int nch = DDIM / KC;
  for (int c = 0; c < nch; ++c) {
    int buf = c & 1;
    wait_async();
    __syncthreads();
    if (c + 1 < nch) stage((c + 1) * KC, buf ^ 1);
#pragma unroll
    for (int ks = 0; ks < KC; ks += 32) {
      v16bf afr[2], bfr[2];
#pragma unroll
      for (int i = 0; i < 2; ++i) {
        const __bf16* ap = &As[buf][(wm * 32 + i * 16 + laneLow) * LDSS + ks + hiSel * 8];
        afr[i] = concat8(*(const v8bf*)ap, *(const v8bf*)(ap + 16));
      }
#pragma unroll
      for (int j = 0; j < 2; ++j) {
        const __bf16* bp = &Bs[buf][(wn * 32 + j * 16 + laneLow) * LDSS + ks + hiSel * 8];
        bfr[j] = concat8(*(const v8bf*)bp, *(const v8bf*)(bp + 16));
      }
#pragma unroll
      for (int i = 0; i < 2; ++i)
#pragma unroll
        for (int j = 0; j < 2; ++j)
          acc[i][j] = __builtin_amdgcn_wmma_f32_16x16x32_bf16(
              false, afr[i], false, bfr[j], (short)0, acc[i][j], false, false);
    }
  }
  const float* bias = b1 + (size_t)e * HDIM;
#pragma unroll
  for (int i = 0; i < 2; ++i)
#pragma unroll
    for (int j = 0; j < 2; ++j) {
      int col = n0 + wn * 32 + j * 16 + laneLow;
      float bv = bias[col];
#pragma unroll
      for (int r = 0; r < 8; ++r) {
        int lr = m0 + wm * 32 + i * 16 + hiSel * 8 + r;
        if (lr < cnt) {
          float v = acc[i][j][r] + bv;
          Hb[(size_t)(gbase + lr) * HDIM + col] = (__bf16)(v > 0.f ? v : 0.f);
        }
      }
    }
}

// -------- GEMM2: Y = (H @ W2_e + b2) * gate_w  (pre-scaled, f32 out) -------
// W2T: [E][D][H] (pre-transposed).
__global__ void __launch_bounds__(256)
ffn2_k(const __bf16* __restrict__ Hb, const __bf16* __restrict__ W2T,
       const float* __restrict__ b2, const __bf16* __restrict__ zrow,
       const float* __restrict__ wlists, const int* __restrict__ counts,
       const int* __restrict__ offs, float* __restrict__ Yg) {
  int e  = blockIdx.z;
  int n0 = blockIdx.x * BN;
  int m0 = blockIdx.y * BM;
  int cnt = counts[e];
  if (m0 >= cnt) return;
  int gbase = offs[e];
  const __bf16* Wt = W2T + (size_t)e * HDIM * DDIM;

  __shared__ __align__(16) __bf16 As[2][BM * LDSS];
  __shared__ __align__(16) __bf16 Bs[2][BN * LDSS];
  __shared__ const __bf16* rowptr[BM];
  __shared__ float gws[BM];

  int tid = threadIdx.x;
  if (tid < BM) {
    int lr = m0 + tid;
    bool valid = lr < cnt;
    rowptr[tid] = valid ? (Hb + (size_t)(gbase + lr) * HDIM) : zrow;
    gws[tid] = valid ? wlists[e * B_TOK + lr] : 0.f;
  }
  __syncthreads();

  int wave = tid >> 5, lane = tid & 31;
  int wm = wave >> 2, wn = wave & 3;
  int laneLow = lane & 15, hiSel = lane >> 4;

  auto stage = [&](int k0, int buf) {
#pragma unroll
    for (int it = 0; it < 2; ++it) {
      int u = tid + it * 256;
      int r = u >> 3, s = u & 7;
      copy16(rowptr[r] + k0 + s * 8, &As[buf][r * LDSS + s * 8]);
    }
#pragma unroll
    for (int it = 0; it < 4; ++it) {
      int u = tid + it * 256;
      int nn = u >> 3, ss = u & 7;
      copy16(Wt + (size_t)(n0 + nn) * HDIM + k0 + ss * 8, &Bs[buf][nn * LDSS + ss * 8]);
    }
  };

  v8f acc[2][2] = {};
  stage(0, 0);
  const int nch = HDIM / KC;
  for (int c = 0; c < nch; ++c) {
    int buf = c & 1;
    wait_async();
    __syncthreads();
    if (c + 1 < nch) stage((c + 1) * KC, buf ^ 1);
#pragma unroll
    for (int ks = 0; ks < KC; ks += 32) {
      v16bf afr[2], bfr[2];
#pragma unroll
      for (int i = 0; i < 2; ++i) {
        const __bf16* ap = &As[buf][(wm * 32 + i * 16 + laneLow) * LDSS + ks + hiSel * 8];
        afr[i] = concat8(*(const v8bf*)ap, *(const v8bf*)(ap + 16));
      }
#pragma unroll
      for (int j = 0; j < 2; ++j) {
        const __bf16* bp = &Bs[buf][(wn * 32 + j * 16 + laneLow) * LDSS + ks + hiSel * 8];
        bfr[j] = concat8(*(const v8bf*)bp, *(const v8bf*)(bp + 16));
      }
#pragma unroll
      for (int i = 0; i < 2; ++i)
#pragma unroll
        for (int j = 0; j < 2; ++j)
          acc[i][j] = __builtin_amdgcn_wmma_f32_16x16x32_bf16(
              false, afr[i], false, bfr[j], (short)0, acc[i][j], false, false);
    }
  }
  const float* bias = b2 + (size_t)e * DDIM;
#pragma unroll
  for (int i = 0; i < 2; ++i)
#pragma unroll
    for (int j = 0; j < 2; ++j) {
      int col = n0 + wn * 32 + j * 16 + laneLow;
      float bv = bias[col];
#pragma unroll
      for (int r = 0; r < 8; ++r) {
        int lrl = wm * 32 + i * 16 + hiSel * 8 + r;
        int lr = m0 + lrl;
        if (lr < cnt) {
          float v = (acc[i][j][r] + bv) * gws[lrl];
          Yg[(size_t)(gbase + lr) * DDIM + col] = v;
        }
      }
    }
}

// ------------------------------ combine ------------------------------------
__global__ void combine_k(const float* __restrict__ Yg, const int* __restrict__ topk_idx,
                          const int* __restrict__ rowmap, const int* __restrict__ offs,
                          float* __restrict__ out) {
  int t = blockIdx.x;
  size_t g0 = (size_t)(offs[topk_idx[t * 2 + 0]] + rowmap[t * 2 + 0]);
  size_t g1 = (size_t)(offs[topk_idx[t * 2 + 1]] + rowmap[t * 2 + 1]);
  const float4* a = (const float4*)(Yg + g0 * DDIM);
  const float4* b = (const float4*)(Yg + g1 * DDIM);
  float4* o = (float4*)(out + (size_t)t * DDIM);
  for (int i = threadIdx.x; i < DDIM / 4; i += blockDim.x) {
    float4 va = a[i], vb = b[i];
    o[i] = make_float4(va.x + vb.x, va.y + vb.y, va.z + vb.z, va.w + vb.w);
  }
}

// ------------------------------ launcher -----------------------------------
extern "C" void kernel_launch(void* const* d_in, const int* in_sizes, int n_in,
                              void* d_out, int out_size, void* d_ws, size_t ws_size,
                              hipStream_t stream) {
  (void)in_sizes; (void)n_in; (void)out_size; (void)ws_size;
  const float* x  = (const float*)d_in[0];
  const float* Wg = (const float*)d_in[1];
  const float* bg = (const float*)d_in[2];
  const float* W1 = (const float*)d_in[3];
  const float* b1 = (const float*)d_in[4];
  const float* W2 = (const float*)d_in[5];
  const float* b2 = (const float*)d_in[6];
  float* out = (float*)d_out;

  char* p = (char*)d_ws;
  auto bump = [&](size_t bytes) -> void* {
    void* r = (void*)p;
    p += (bytes + 255) & ~(size_t)255;
    return r;
  };
  __bf16* xb   = (__bf16*)bump((size_t)B_TOK * DDIM * 2);
  __bf16* W1T  = (__bf16*)bump((size_t)NEXP * DDIM * HDIM * 2);
  __bf16* W2T  = (__bf16*)bump((size_t)NEXP * HDIM * DDIM * 2);
  __bf16* Hb   = (__bf16*)bump((size_t)B_TOK * 2 * HDIM * 2);
  float*  Yg   = (float*) bump((size_t)B_TOK * 2 * DDIM * 4);
  __bf16* zrow = (__bf16*)bump((size_t)HDIM * 2);
  int*    tidx = (int*)   bump((size_t)B_TOK * 2 * 4);
  float*  tw   = (float*) bump((size_t)B_TOK * 2 * 4);
  int*    lists  = (int*)   bump((size_t)NEXP * B_TOK * 4);
  float*  wlists = (float*) bump((size_t)NEXP * B_TOK * 4);
  int*    rowmap = (int*)   bump((size_t)B_TOK * 2 * 4);
  int*    counts = (int*)   bump(64);
  int*    offs   = (int*)   bump(64);
  float*  partial = (float*)bump((size_t)1024 * NEXP * 4);

  // 1) bf16 conversion (x) + transposing conversions (W1 -> [E,H,D], W2 -> [E,D,H])
  cvt_bf16_k<<<(B_TOK * DDIM / 4) / 256, 256, 0, stream>>>(x, xb);
  cvt_tr_k<<<dim3(HDIM / 64, DDIM / 64, NEXP), 256, 0, stream>>>(W1, W1T, DDIM, HDIM);
  cvt_tr_k<<<dim3(DDIM / 64, HDIM / 64, NEXP), 256, 0, stream>>>(W2, W2T, HDIM, DDIM);
  zero_bf16_k<<<2, 256, 0, stream>>>(zrow);

  // 2) gating + aux
  gating_k<<<B_TOK / 8, 256, 0, stream>>>(x, Wg, bg, tidx, tw, partial);
  aux_k<<<1, 32, 0, stream>>>(partial, B_TOK / 8, out + (size_t)B_TOK * DDIM);

  // 3) routing
  compact_k<<<NEXP, 256, 0, stream>>>(tidx, tw, lists, wlists, rowmap, counts);
  offsets_k<<<1, 32, 0, stream>>>(counts, offs);

  // 4) expert FFN (grouped GEMMs, bf16 WMMA, double-buffered async LDS staging)
  dim3 g1(HDIM / BN, B_TOK / BM, NEXP);
  ffn1_k<<<g1, 256, 0, stream>>>(xb, W1T, b1, zrow, lists, counts, offs, Hb);
  dim3 g2(DDIM / BN, B_TOK / BM, NEXP);
  ffn2_k<<<g2, 256, 0, stream>>>(Hb, W2T, b2, zrow, wlists, counts, offs, Yg);

  // 5) weighted combine
  combine_k<<<B_TOK, 256, 0, stream>>>(Yg, tidx, rowmap, offs, out);
}
